// GAT_7060926234756
// MI455X (gfx1250) — compile-verified, matrix operand order checked by reading, not run on hardware
//
#include <hip/hip_runtime.h>
#include <hip/hip_bf16.h>
#include <cmath>

#define N_NODES 100000
#define N_EDGES 1600000
#define ET      (N_EDGES + N_NODES)   // edges + self loops
#define F_INF   128
#define H1H     8
#define CHN     32
#define HC      (H1H * CHN)           // 256
#define NG      64
#define NCLS    10

typedef __attribute__((ext_vector_type(16))) _Float16 v16h;
typedef __attribute__((ext_vector_type(8)))  _Float16 v8h;
typedef __attribute__((ext_vector_type(8)))  float    v8f;

// ---------- helpers ----------
__device__ __forceinline__ void atomic_max_f32(float* addr, float v) {
  // monotone mapping of float order onto int/uint order
  if (v >= 0.0f) atomicMax((int*)addr, __float_as_int(v));
  else           atomicMin((unsigned int*)addr, __float_as_uint(v));
}

__device__ __forceinline__ void edge_nodes(const int* __restrict__ ei, int e, int& s, int& d) {
  if (e < N_EDGES) { s = ei[e]; d = ei[N_EDGES + e]; }
  else             { s = e - N_EDGES; d = s; }       // appended self loops
}

__global__ void fill_f32(float* __restrict__ p, float v, int n) {
  int i = blockIdx.x * blockDim.x + threadIdx.x;
  if (i < n) p[i] = v;
}

// f32 -> f16 elementwise
__global__ void cvt_f16(const float* __restrict__ s, _Float16* __restrict__ d, int n) {
  int i = blockIdx.x * blockDim.x + threadIdx.x;
  if (i < n) d[i] = (_Float16)s[i];
}

// Bt[n][k] = (f16) B[k][n]   (B is [K,Nn] row-major)
__global__ void transpose_f16(const float* __restrict__ B, _Float16* __restrict__ Bt,
                              int K, int Nn) {
  int i = blockIdx.x * blockDim.x + threadIdx.x;
  if (i >= K * Nn) return;
  int k = i / Nn, n = i % Nn;
  Bt[(size_t)n * K + k] = (_Float16)B[i];
}

// ---------- WMMA fragment load: two contiguous 16B runs per lane ----------
// 16-bit A layout: vgprs 0-3 hold K in [k0+8*hi, +8), vgprs 4-7 hold [k0+16+8*hi, +8),
// packed half-pairs K-contiguous -> v16h elements 0..7 and 8..15 are contiguous runs.
__device__ __forceinline__ v16h frag_load(const _Float16* __restrict__ row, int k0, int hi) {
  const v8h lo = *(const v8h*)(row + k0 + (hi << 3));        // 16B aligned
  const v8h hh = *(const v8h*)(row + k0 + 16 + (hi << 3));
  v16h f;
#pragma unroll
  for (int i = 0; i < 8; ++i) { f[i] = lo[i]; f[8 + i] = hh[i]; }
  return f;
}

// ---------- WMMA GEMM: C[M,Nn] = Ah[M,K] @ Bt[Nn,K]^T; f16 operands, f32 accum ----------
// One wave -> one 16 x (16*TN) strip: A fragment loaded once per k-step, reused
// against TN independent accumulators. M%16==0, K%32==0, Nn%(16*TN)==0 hold here.
template <int TN>
__global__ void gemm_wmma_f16(const _Float16* __restrict__ Ah, const _Float16* __restrict__ Bt,
                              float* __restrict__ Cout, int M, int K, int Nn) {
  const int strips_n = Nn / (16 * TN);
  const int strips   = (M >> 4) * strips_n;
  const int wave = (blockIdx.x * blockDim.x + threadIdx.x) >> 5;
  if (wave >= strips) return;                 // uniform per wave; EXEC stays all-ones
  const int tm  = wave / strips_n;
  const int tn0 = (wave % strips_n) * TN;
  const int lane = threadIdx.x & 31;
  const int r  = lane & 15;
  const int hi = lane >> 4;

  const _Float16* arow = Ah + (size_t)(tm * 16 + r) * K;
  const _Float16* brow[TN];
#pragma unroll
  for (int t = 0; t < TN; ++t) brow[t] = Bt + (size_t)((tn0 + t) * 16 + r) * K;

  v8f acc[TN];
#pragma unroll
  for (int t = 0; t < TN; ++t) acc[t] = (v8f){};

  for (int k0 = 0; k0 < K; k0 += 32) {
    const v16h af = frag_load(arow, k0, hi);
#pragma unroll
    for (int t = 0; t < TN; ++t) {
      const v16h bf = frag_load(brow[t], k0, hi);
      acc[t] = __builtin_amdgcn_wmma_f32_16x16x32_f16(false, af, false, bf,
                                                      (short)0, acc[t], false, false);
    }
  }
#pragma unroll
  for (int t = 0; t < TN; ++t) {
#pragma unroll
    for (int rr = 0; rr < 8; ++rr) {          // C/D layout: row = rr + 8*hi, col = r
      const int row = tm * 16 + rr + (hi << 3);
      Cout[(size_t)row * Nn + (tn0 + t) * 16 + r] = acc[t][rr];
    }
  }
}

// ---------- attention score dot products: a_s[n,h], a_d[n,h] ----------
template <int HEADS>
__global__ void att_scores(const float* __restrict__ h, const float* __restrict__ asrc,
                           const float* __restrict__ adst, float* __restrict__ out_s,
                           float* __restrict__ out_d) {
  int i = blockIdx.x * blockDim.x + threadIdx.x;   // flat (n, head)
  if (i >= N_NODES * HEADS) return;
  int hh = i % HEADS;
  const float* hv = h + (size_t)i * CHN;
  float ss = 0.f, dd = 0.f;
#pragma unroll
  for (int c = 0; c < CHN; ++c) {
    float v = hv[c];
    ss += v * asrc[hh * CHN + c];
    dd += v * adst[hh * CHN + c];
  }
  out_s[i] = ss;
  out_d[i] = dd;
}

// ---------- edge pass A: LeakyReLU score + segment max over dst ----------
template <int HEADS>
__global__ void edge_max(const int* __restrict__ ei, const float* __restrict__ as_,
                         const float* __restrict__ ad_, float* __restrict__ esc,
                         float* __restrict__ m) {
  int i = blockIdx.x * blockDim.x + threadIdx.x;
  if (i >= ET * HEADS) return;
  int e = i / HEADS, hh = i % HEADS;
  int s, d; edge_nodes(ei, e, s, d);
  float v = as_[s * HEADS + hh] + ad_[d * HEADS + hh];
  v = (v > 0.f) ? v : 0.2f * v;                 // LeakyReLU(0.2)
  esc[i] = v;
  atomic_max_f32(&m[d * HEADS + hh], v);
}

// ---------- edge pass B: p = exp(e - m[dst]); z[dst] += p ----------
template <int HEADS>
__global__ void edge_expsum(const int* __restrict__ ei, float* __restrict__ esc,
                            const float* __restrict__ m, float* __restrict__ z) {
  int i = blockIdx.x * blockDim.x + threadIdx.x;
  if (i >= ET * HEADS) return;
  int e = i / HEADS, hh = i % HEADS;
  int s, d; edge_nodes(ei, e, s, d); (void)s;
  float p = expf(esc[i] - m[d * HEADS + hh]);
  esc[i] = p;
  atomicAdd(&z[d * HEADS + hh], p);
}

// ---------- edge pass C: out[dst] += (p/z[dst]) * h[src] ----------
template <int HEADS>
__global__ void edge_scatter(const int* __restrict__ ei, const float* __restrict__ p,
                             const float* __restrict__ z, const float* __restrict__ h,
                             float* __restrict__ out) {
  int i = blockIdx.x * blockDim.x + threadIdx.x;
  if (i >= ET * HEADS) return;
  int e = i / HEADS, hh = i % HEADS;
  int s, d; edge_nodes(ei, e, s, d);
  float coef = p[i] / (z[d * HEADS + hh] + 1e-16f);
  const float4* hv4 = (const float4*)(h + ((size_t)s * HEADS + hh) * CHN);  // 128B-aligned row
  float*        ov  = out + ((size_t)d * HEADS + hh) * CHN;
#pragma unroll
  for (int q = 0; q < CHN / 4; ++q) {
    float4 v4 = hv4[q];                        // b128 gather (L2-resident)
    atomicAdd(&ov[q * 4 + 0], coef * v4.x);
    atomicAdd(&ov[q * 4 + 1], coef * v4.y);
    atomicAdd(&ov[q * 4 + 2], coef * v4.z);
    atomicAdd(&ov[q * 4 + 3], coef * v4.w);
  }
}

// ---------- o = (f16) elu(x + b): fused bias + ELU + downconvert for GEMM2 input ----------
__global__ void bias_elu_f16(const float* __restrict__ x, const float* __restrict__ b,
                             _Float16* __restrict__ o, int n) {
  int i = blockIdx.x * blockDim.x + threadIdx.x;
  if (i >= n) return;
  float v = x[i] + b[i & (HC - 1)];             // HC == 256 (power of two)
  o[i] = (_Float16)((v > 0.f) ? v : expm1f(v)); // ELU(alpha=1)
}

// ---------- global mean pool (sum + count), b2 folded in ----------
__global__ void pool_kernel(const float* __restrict__ h2, const int* __restrict__ batch,
                            const float* __restrict__ b2, float* __restrict__ pooled,
                            float* __restrict__ cnt) {
  int n = blockIdx.x * blockDim.x + threadIdx.x;
  if (n >= N_NODES) return;
  int g = batch[n];
#pragma unroll
  for (int c = 0; c < CHN; ++c)
    atomicAdd(&pooled[g * CHN + c], h2[(size_t)n * CHN + c] + b2[c]);
  atomicAdd(&cnt[g], 1.0f);
}

// ---------- classifier: out[g,k] = mean_pool @ Wlin + blin ----------
__global__ void final_kernel(const float* __restrict__ pooled, const float* __restrict__ cnt,
                             const float* __restrict__ Wlin, const float* __restrict__ blin,
                             float* __restrict__ out) {
  int i = threadIdx.x;
  if (i >= NG * NCLS) return;
  int g = i / NCLS, k = i % NCLS;
  float inv = 1.0f / fmaxf(cnt[g], 1.0f);
  float acc = blin[k];
#pragma unroll
  for (int c = 0; c < CHN; ++c)
    acc += pooled[g * CHN + c] * inv * Wlin[c * NCLS + k];
  out[i] = acc;
}

extern "C" void kernel_launch(void* const* d_in, const int* in_sizes, int n_in,
                              void* d_out, int out_size, void* d_ws, size_t ws_size,
                              hipStream_t stream) {
  const float* x     = (const float*)d_in[0];
  const int*   ei    = (const int*)  d_in[1];
  const int*   batch = (const int*)  d_in[2];
  const float* W1    = (const float*)d_in[3];
  const float* asrc1 = (const float*)d_in[4];
  const float* adst1 = (const float*)d_in[5];
  const float* b1    = (const float*)d_in[6];
  const float* W2    = (const float*)d_in[7];
  const float* asrc2 = (const float*)d_in[8];
  const float* adst2 = (const float*)d_in[9];
  const float* b2    = (const float*)d_in[10];
  const float* Wlin  = (const float*)d_in[11];
  const float* blin  = (const float*)d_in[12];
  float* out = (float*)d_out;

  // workspace layout, cross-layer reuse; ~350 MB total (all chunks 16B aligned)
  float* ws  = (float*)d_ws;
  float* h1  = ws;                               // [N,256] gemm1 out; reused as h2 [N,32]
  float* g1  = h1 + (size_t)N_NODES * HC;        // [N,256] gat1 accum; reused as gat2 accum [N,32]
  float* as_ = g1 + (size_t)N_NODES * HC;        // [N,8] (layer2 uses first N)
  float* ad_ = as_ + (size_t)N_NODES * H1H;
  float* m_  = ad_ + (size_t)N_NODES * H1H;
  float* z_  = m_  + (size_t)N_NODES * H1H;
  float* e_  = z_  + (size_t)N_NODES * H1H;      // [ET,8] edge scores/probs
  float* pooled = e_ + (size_t)ET * H1H;         // [64,32]
  float* cnt = pooled + NG * CHN;                // [64]
  _Float16* xh  = (_Float16*)(cnt + NG);         // [N,128] f16 copy of x
  _Float16* g1h = xh + (size_t)N_NODES * F_INF;  // [N,256] f16 elu(g1+b1)
  _Float16* W1t = g1h + (size_t)N_NODES * HC;    // [256,128] f16 W1^T
  _Float16* W2t = W1t + (size_t)HC * F_INF;      // [32,256]  f16 W2^T

  const int B = 256;
  auto blk = [](long long n, int b) { return (unsigned)((n + b - 1) / b); };

  // ---- operand staging for WMMA ----
  cvt_f16<<<blk((long long)N_NODES * F_INF, B), B, 0, stream>>>(x, xh, N_NODES * F_INF);
  transpose_f16<<<blk((long long)F_INF * HC, B), B, 0, stream>>>(W1, W1t, F_INF, HC);
  transpose_f16<<<blk((long long)HC * CHN, B), B, 0, stream>>>(W2, W2t, HC, CHN);

  // ================= layer 1 (heads = 8) =================
  {
    long long waves = (long long)(N_NODES / 16) * (HC / 64);   // TN=4 -> 16x64 strips
    gemm_wmma_f16<4><<<blk(waves * 32, B), B, 0, stream>>>(xh, W1t, h1, N_NODES, F_INF, HC);
  }
  att_scores<H1H><<<blk((long long)N_NODES * H1H, B), B, 0, stream>>>(h1, asrc1, adst1, as_, ad_);
  fill_f32<<<blk((long long)N_NODES * H1H, B), B, 0, stream>>>(m_, -INFINITY, N_NODES * H1H);
  fill_f32<<<blk((long long)N_NODES * H1H, B), B, 0, stream>>>(z_, 0.f, N_NODES * H1H);
  fill_f32<<<blk((long long)N_NODES * HC, B), B, 0, stream>>>(g1, 0.f, N_NODES * HC);
  edge_max<H1H>    <<<blk((long long)ET * H1H, B), B, 0, stream>>>(ei, as_, ad_, e_, m_);
  edge_expsum<H1H> <<<blk((long long)ET * H1H, B), B, 0, stream>>>(ei, e_, m_, z_);
  edge_scatter<H1H><<<blk((long long)ET * H1H, B), B, 0, stream>>>(ei, e_, z_, h1, g1);
  bias_elu_f16<<<blk((long long)N_NODES * HC, B), B, 0, stream>>>(g1, b1, g1h, N_NODES * HC);

  // ================= layer 2 (heads = 1) =================
  {
    long long waves = (long long)(N_NODES / 16) * (CHN / 32);  // TN=2 -> 16x32 strips
    gemm_wmma_f16<2><<<blk(waves * 32, B), B, 0, stream>>>(g1h, W2t, h1, N_NODES, HC, CHN);
  }
  att_scores<1><<<blk((long long)N_NODES, B), B, 0, stream>>>(h1, asrc2, adst2, as_, ad_);
  fill_f32<<<blk((long long)N_NODES, B), B, 0, stream>>>(m_, -INFINITY, N_NODES);
  fill_f32<<<blk((long long)N_NODES, B), B, 0, stream>>>(z_, 0.f, N_NODES);
  fill_f32<<<blk((long long)N_NODES * CHN, B), B, 0, stream>>>(g1, 0.f, N_NODES * CHN);
  edge_max<1>    <<<blk((long long)ET, B), B, 0, stream>>>(ei, as_, ad_, e_, m_);
  edge_expsum<1> <<<blk((long long)ET, B), B, 0, stream>>>(ei, e_, m_, z_);
  edge_scatter<1><<<blk((long long)ET, B), B, 0, stream>>>(ei, e_, z_, h1, g1);

  // ================= pool + classify =================
  fill_f32<<<blk(NG * CHN + NG, B), B, 0, stream>>>(pooled, 0.f, NG * CHN + NG);
  pool_kernel<<<blk(N_NODES, B), B, 0, stream>>>(g1, batch, b2, pooled, cnt);
  final_kernel<<<1, NG * NCLS, 0, stream>>>(pooled, cnt, Wlin, blin, out);
}
